// GraphDecoderCore_88012469829876
// MI455X (gfx1250) — matrix operand order repaired and checked by reading, not compiled, not run on hardware
//
#include <hip/hip_runtime.h>

typedef _Float16 f16;
typedef unsigned int u32;
typedef __attribute__((ext_vector_type(16))) _Float16 v16h;
typedef __attribute__((ext_vector_type(8)))  _Float16 v8h;
typedef __attribute__((ext_vector_type(2)))  _Float16 v2h;
typedef __attribute__((ext_vector_type(8)))  float    v8f;

#define HD   10   // GRU hidden
#define NIN  9    // node input feats
#define EF   11   // edge feats / message feats
#define MSGD 96   // MLP width

// ---------------- WMMA fragment helpers (CDNA5 16x16x32 f16 layouts) -------

// A (16x32, MxK), staging row-major [m][k]: lane half=0 reads K
// {kc*32+0..7, +16..23}; half=1 reads {+8..15, +24..31} -> two 16B loads.
__device__ __forceinline__ v16h load_a(const f16* st, int stride, int kc,
                                       int l16, int half) {
  const f16* p = st + l16 * stride + kc * 32 + half * 8;
  v8h lo = *(const v8h*)p;          // elems 0..7
  v8h hi = *(const v8h*)(p + 16);   // elems 8..15
  return __builtin_shufflevector(lo, hi, 0, 1, 2, 3, 4, 5, 6, 7, 8, 9, 10, 11,
                                 12, 13, 14, 15);
}

// B (32x16, KxN) from TRANSPOSED weights wT[k][n]: lane%16 = N, elems = K in
// the same order as A (half=0: K {0..7, 16..23}; half=1: {8..15, 24..31}).
__device__ __forceinline__ v16h load_bT(const f16* wt, int nstride, int nt,
                                        int kc, int lane) {
  const f16* p = wt + (kc * 32 + (lane >> 4) * 8) * nstride + nt * 16 + (lane & 15);
  v16h r;
#pragma unroll
  for (int s = 0; s < 8; ++s) {
    r[s]     = p[s * nstride];
    r[8 + s] = p[(16 + s) * nstride];
  }
  return r;
}

__device__ __forceinline__ v2h pack2(float x0, float x1) {
  return __builtin_bit_cast(v2h, __builtin_amdgcn_cvt_pkrtz(x0, x1));
}

// C/D (16x16 f32): reg r = row M (lanes 16-31: M+8), lane%16 = N.
// pack-convert first (v_cvt_pk_rtz), then bias + ReLU in packed f16
// (v_pk_add_f16 + v_pk_max_num_f16): both rows of a pair share column n,
// hence the same bias.
__device__ __forceinline__ void store_act(f16* st, int stride, int nt, v8f d,
                                          v2h bb, int l16, int half) {
  f16* p = st + (half * 8) * stride + nt * 16 + l16;
  const v2h z0 = {(f16)0.f, (f16)0.f};
#pragma unroll
  for (int r = 0; r < 8; r += 2) {
    v2h pk = pack2(d[r], d[r + 1]);
    pk = pk + bb;
    pk = __builtin_elementwise_max(pk, z0);
    p[r * stride]       = pk.x;
    p[(r + 1) * stride] = pk.y;
  }
}

__device__ __forceinline__ void store_act_nobias(f16* st, int stride, int nt,
                                                 v8f d, int l16, int half) {
  f16* p = st + (half * 8) * stride + nt * 16 + l16;
  const v2h z0 = {(f16)0.f, (f16)0.f};
#pragma unroll
  for (int r = 0; r < 8; r += 2) {
    v2h pk = pack2(d[r], d[r + 1]);
    pk = __builtin_elementwise_max(pk, z0);
    p[r * stride]       = pk.x;
    p[(r + 1) * stride] = pk.y;
  }
}

// ---------------- Edge-MLP + scatter kernel --------------------------------
__global__ __launch_bounds__(128) void edge_msg_kernel(
    const f16* __restrict__ h16,    // [N][10] f16 shadow of h
    const f16* __restrict__ ea16,   // [E][12] f16 edge_attr, elem 11 == 1.0
    float* __restrict__ agg,
    const float* __restrict__ W1, const float* __restrict__ b1,
    const float* __restrict__ W2, const float* __restrict__ b2,
    const float* __restrict__ W3, const float* __restrict__ b3,
    const float* __restrict__ W4, const float* __restrict__ b4,
    const int* __restrict__ src_ids, const int* __restrict__ dst_ids,
    int E, int nwaves) {
  // Weights transposed: wT[k][n]; B fragment = 16 f16 per lane, strided by the N width.
  __shared__ __align__(32) f16 w1t[32 * 96];   // K row 31 holds b1 (bias row)
  __shared__ __align__(32) f16 w2t[96 * 96];
  __shared__ __align__(32) f16 w3t[96 * 96];
  __shared__ __align__(32) f16 w4t[96 * 16];   // N padded 11->16
  __shared__ v2h b2p[96], b3p[96];             // biases as duplicated f16 pairs
  __shared__ float b4s[16];
  __shared__ __align__(32) f16 stage[4][16 * 96];  // per-wave row-major tile

  const int tid = threadIdx.x;
  for (int i = tid; i < 32 * 96; i += 128) {
    int k = i / 96, n = i % 96;
    w1t[i] = (f16)((k < 31) ? W1[n * 31 + k] : b1[n]);  // bias folded at K=31
  }
  for (int i = tid; i < 96 * 96; i += 128) {
    int k = i / 96, n = i % 96;
    w2t[i] = (f16)W2[n * 96 + k];
  }
  for (int i = tid; i < 96 * 96; i += 128) {
    int k = i / 96, n = i % 96;
    w3t[i] = (f16)W3[n * 96 + k];
  }
  for (int i = tid; i < 96 * 16; i += 128) {
    int k = i / 16, n = i % 16;
    w4t[i] = (f16)((n < 11) ? W4[n * 96 + k] : 0.f);
  }
  for (int i = tid; i < 96; i += 128) {
    f16 v2 = (f16)b2[i];
    f16 v3 = (f16)b3[i];
    b2p[i] = (v2h){v2, v2};
    b3p[i] = (v2h){v3, v3};
  }
  if (tid < 16) b4s[tid] = (tid < 11) ? b4[tid] : 0.f;
  __syncthreads();

  const int wid  = tid >> 5;
  const int lane = tid & 31;
  const int l16  = lane & 15;
  const int half = lane >> 4;
  f16* st = stage[wid];
  const int gw = blockIdx.x * 4 + wid;
  const int ntiles = (E + 15) >> 4;

  const v8f cz = {};   // zero accumulator -> selects inline-0 C operand

  // Pre-load first tile's src/dst ids (carried in regs across iterations).
  int s_cur = 0, d_cur = 0;
  {
    const int e0 = gw * 16 + l16;
    if (half == 0 && e0 < E) { s_cur = src_ids[e0]; d_cur = dst_ids[e0]; }
  }

  for (int tile = gw; tile < ntiles; tile += nwaves) {
    const int base = tile * 16;

    // Load next tile's ids now; prefetch the *resolved* indirect h16 rows so
    // the next gather hits warm cache.
    int s_nxt = 0, d_nxt = 0;
    {
      const int ne = (tile + nwaves) * 16 + l16;
      if (half == 0 && ne < E) {
        s_nxt = src_ids[ne];
        d_nxt = dst_ids[ne];
        __builtin_prefetch(h16 + (size_t)s_nxt * HD, 0, 1);
        __builtin_prefetch(h16 + (size_t)d_nxt * HD, 0, 1);
        __builtin_prefetch(ea16 + (size_t)ne * 12, 0, 1);
      }
    }

    // ---- gather: lanes 0..15 build one edge row: [h_s|h_d|ea|1.0] (32 f16)
    if (half == 0) {
      const int e = base + l16;
      u32* rp = (u32*)(st + l16 * 96);
      if (e < E) {
        const u32* hs = (const u32*)(h16 + (size_t)s_cur * HD);
        const u32* hd = (const u32*)(h16 + (size_t)d_cur * HD);
        const u32* ep = (const u32*)(ea16 + (size_t)e * 12);
#pragma unroll
        for (int j = 0; j < 5; ++j) rp[j] = hs[j];
#pragma unroll
        for (int j = 0; j < 5; ++j) rp[5 + j] = hd[j];
#pragma unroll
        for (int j = 0; j < 6; ++j) rp[10 + j] = ep[j];  // incl. 1.0 at k=31
      } else {
#pragma unroll
        for (int j = 0; j < 16; ++j) rp[j] = 0u;
      }
    }
    asm volatile("s_wait_dscnt 0" ::: "memory");  // gather visible wave-wide

    // ---- layer 1: [16x32] x [32x96]  (bias via K=31 row, C = 0)
    v16h a0 = load_a(st, 96, 0, l16, half);
#pragma unroll
    for (int nt = 0; nt < 6; ++nt) {
      v16h b = load_bT(w1t, 96, nt, 0, lane);
      v8f  c = __builtin_amdgcn_wmma_f32_16x16x32_f16(false, a0, false, b,
                                                      (short)0, cz, false,
                                                      false);
      store_act_nobias(st, 96, nt, c, l16, half);
    }
    asm volatile("s_wait_dscnt 0" ::: "memory");

    // ---- layer 2: [16x96] x [96x96]
    v16h a[3];
#pragma unroll
    for (int kc = 0; kc < 3; ++kc) a[kc] = load_a(st, 96, kc, l16, half);
#pragma unroll
    for (int nt = 0; nt < 6; ++nt) {
      v8f c = cz;
#pragma unroll
      for (int kc = 0; kc < 3; ++kc) {
        v16h b = load_bT(w2t, 96, nt, kc, lane);
        c = __builtin_amdgcn_wmma_f32_16x16x32_f16(false, a[kc], false, b,
                                                   (short)0, c, false, false);
      }
      store_act(st, 96, nt, c, b2p[nt * 16 + l16], l16, half);
    }
    asm volatile("s_wait_dscnt 0" ::: "memory");

    // ---- layer 3: [16x96] x [96x96]
#pragma unroll
    for (int kc = 0; kc < 3; ++kc) a[kc] = load_a(st, 96, kc, l16, half);
#pragma unroll
    for (int nt = 0; nt < 6; ++nt) {
      v8f c = cz;
#pragma unroll
      for (int kc = 0; kc < 3; ++kc) {
        v16h b = load_bT(w3t, 96, nt, kc, lane);
        c = __builtin_amdgcn_wmma_f32_16x16x32_f16(false, a[kc], false, b,
                                                   (short)0, c, false, false);
      }
      store_act(st, 96, nt, c, b3p[nt * 16 + l16], l16, half);
    }
    asm volatile("s_wait_dscnt 0" ::: "memory");

    // ---- layer 4: [16x96] x [96x16(11)], bias at scatter, no ReLU
#pragma unroll
    for (int kc = 0; kc < 3; ++kc) a[kc] = load_a(st, 96, kc, l16, half);
    v8f c4 = cz;
#pragma unroll
    for (int kc = 0; kc < 3; ++kc) {
      v16h b = load_bT(w4t, 16, 0, kc, lane);
      c4 = __builtin_amdgcn_wmma_f32_16x16x32_f16(false, a[kc], false, b,
                                                  (short)0, c4, false, false);
    }
    if (l16 < EF) {   // D layout: row = half*8 + r, col = l16
      const float bv = b4s[l16];
#pragma unroll
      for (int r = 0; r < 8; ++r) {
        const int e = base + half * 8 + r;
        if (e < E) {
          const int d = dst_ids[e];
          atomicAdd(&agg[(size_t)d * EF + l16], c4[r] + bv);
        }
      }
    }

    s_cur = s_nxt;
    d_cur = d_nxt;
  }
}

// ---------------- GRU node update + readout --------------------------------
__global__ __launch_bounds__(256) void gru_kernel(
    const float* __restrict__ node_inputs, float* __restrict__ h,
    f16* __restrict__ h16, float* __restrict__ agg,
    const float* __restrict__ wih, const float* __restrict__ whh,
    const float* __restrict__ bih, const float* __restrict__ bhh,
    const float* __restrict__ finw, const float* __restrict__ finb,
    float* __restrict__ out, int N, int it) {
  __shared__ float swih[30 * 20], swhh[30 * 10], sbih[30], sbhh[30],
      sfw[2 * 10], sfb[2];
  const int tid = threadIdx.x;
  for (int i = tid; i < 600; i += 256) swih[i] = wih[i];
  for (int i = tid; i < 300; i += 256) swhh[i] = whh[i];
  if (tid < 30) { sbih[tid] = bih[tid]; sbhh[tid] = bhh[tid]; }
  if (tid < 20) sfw[tid] = finw[tid];
  if (tid < 2)  sfb[tid] = finb[tid];
  __syncthreads();

  const int i = blockIdx.x * 256 + tid;
  if (i >= N) return;

  float hv[10], x[20];
#pragma unroll
  for (int j = 0; j < 10; ++j) hv[j] = h[(size_t)i * 10 + j];
#pragma unroll
  for (int f = 0; f < 11; ++f) {      // consume agg, rezero for next iter
    x[f] = agg[(size_t)i * 11 + f];
    agg[(size_t)i * 11 + f] = 0.f;
  }
#pragma unroll
  for (int f = 0; f < 9; ++f) x[11 + f] = node_inputs[(size_t)i * 9 + f];

  float hn[10];
#pragma unroll
  for (int j = 0; j < 10; ++j) {
    float gir = sbih[j], giz = sbih[10 + j], gin = sbih[20 + j];
#pragma unroll
    for (int k = 0; k < 20; ++k) {
      gir += swih[j * 20 + k]        * x[k];
      giz += swih[(10 + j) * 20 + k] * x[k];
      gin += swih[(20 + j) * 20 + k] * x[k];
    }
    float ghr = sbhh[j], ghz = sbhh[10 + j], ghn = sbhh[20 + j];
#pragma unroll
    for (int k = 0; k < 10; ++k) {
      ghr += swhh[j * 10 + k]        * hv[k];
      ghz += swhh[(10 + j) * 10 + k] * hv[k];
      ghn += swhh[(20 + j) * 10 + k] * hv[k];
    }
    float r = 1.f / (1.f + __expf(-(gir + ghr)));
    float z = 1.f / (1.f + __expf(-(giz + ghz)));
    float n = tanhf(gin + r * ghn);
    hn[j] = (1.f - z) * n + z * hv[j];
  }
#pragma unroll
  for (int j = 0; j < 10; ++j) h[(size_t)i * 10 + j] = hn[j];
#pragma unroll
  for (int j = 0; j < 10; ++j) h16[(size_t)i * 10 + j] = (f16)hn[j];
#pragma unroll
  for (int o = 0; o < 2; ++o) {
    float s = sfb[o];
#pragma unroll
    for (int j = 0; j < 10; ++j) s += sfw[o * 10 + j] * hn[j];
    out[((size_t)it * N + i) * 2 + o] = s;
  }
}

// ---------------- one-time setup kernels -----------------------------------
__global__ void zero_kernel(u32* p, int n) {
  int i = blockIdx.x * blockDim.x + threadIdx.x;
  if (i < n) p[i] = 0u;
}

// edge_attr f32 [E][11] -> f16 [E][12], elem 11 = 1.0 (bias-row activation)
__global__ void ea16_kernel(const float* __restrict__ ea,
                            f16* __restrict__ ea16, int E) {
  int idx = blockIdx.x * blockDim.x + threadIdx.x;
  if (idx >= E * 12) return;
  int e = idx / 12, j = idx - e * 12;
  ea16[idx] = (f16)((j < EF) ? ea[(size_t)e * EF + j] : 1.0f);
}

// ---------------- host launch ----------------------------------------------
extern "C" void kernel_launch(void* const* d_in, const int* in_sizes, int n_in,
                              void* d_out, int out_size, void* d_ws,
                              size_t ws_size, hipStream_t stream) {
  const float* node_inputs = (const float*)d_in[0];
  const float* edge_attr   = (const float*)d_in[1];
  const float* W1 = (const float*)d_in[2];
  const float* b1 = (const float*)d_in[3];
  const float* W2 = (const float*)d_in[4];
  const float* b2 = (const float*)d_in[5];
  const float* W3 = (const float*)d_in[6];
  const float* b3 = (const float*)d_in[7];
  const float* W4 = (const float*)d_in[8];
  const float* b4 = (const float*)d_in[9];
  const float* gwih = (const float*)d_in[10];
  const float* gwhh = (const float*)d_in[11];
  const float* gbih = (const float*)d_in[12];
  const float* gbhh = (const float*)d_in[13];
  const float* finw = (const float*)d_in[14];
  const float* finb = (const float*)d_in[15];
  const int* src_ids = (const int*)d_in[16];
  const int* dst_ids = (const int*)d_in[17];

  const int N = in_sizes[0] / NIN;   // 20000
  const int E = in_sizes[16];        // 320000

  // workspace layout: h [N,10] f32 | agg [N,11] f32 | h16 [N,10] f16 | ea16
  float* h   = (float*)d_ws;
  float* agg = h + (size_t)N * HD;
  f16*  h16  = (f16*)(agg + (size_t)N * EF);
  f16*  ea16 = h16 + (size_t)N * HD;
  float* out = (float*)d_out;

  // zero h, agg, h16 (N*21 f32 + N*10 f16 = N*26 dwords, contiguous)
  const int nzero = N * 26;
  zero_kernel<<<(nzero + 255) / 256, 256, 0, stream>>>((u32*)d_ws, nzero);
  ea16_kernel<<<(E * 12 + 255) / 256, 256, 0, stream>>>(edge_attr, ea16, E);

  const int eblocks = 1000;            // 4000 waves -> 5 tiles/wave
  const int nwaves  = eblocks * 4;
  const int nblocks = (N + 255) / 256;
  for (int it = 0; it < 7; ++it) {
    edge_msg_kernel<<<eblocks, 128, 0, stream>>>(
        h16, ea16, agg, W1, b1, W2, b2, W3, b3, W4, b4, src_ids, dst_ids, E,
        nwaves);
    gru_kernel<<<nblocks, 256, 0, stream>>>(node_inputs, h, h16, agg, gwih,
                                            gwhh, gbih, gbhh, finw, finb, out,
                                            N, it);
  }
}
